// SelfAttentionHead_73495480369510
// MI455X (gfx1250) — matrix operand order repaired and checked by reading, not compiled, not run on hardware
//
#include <hip/hip_runtime.h>

// ---------------------------------------------------------------------------
// Self-attention head (B=8, T=2048, D=1024, H=64), fp32 in/out.
// Kernel 1: fused Q/K/V projection GEMM (f16 WMMA, f32 accum) + prefetch,
//           V stored transposed [B][H][T].
// Kernel 2: flash attention; per-block TDM (tensor_load_to_lds) double-
//           buffered K/V staging in LDS, v_wmma_f32_16x16x32_f16 for QK^T
//           and PV, online softmax via ds_swizzle reductions.
// ---------------------------------------------------------------------------

typedef _Float16 f16;
typedef f16   v16h __attribute__((ext_vector_type(16)));
typedef f16   v8h  __attribute__((ext_vector_type(8)));
typedef f16   v4h  __attribute__((ext_vector_type(4)));
typedef float v8f  __attribute__((ext_vector_type(8)));
typedef float v4f  __attribute__((ext_vector_type(4)));
typedef unsigned int u32x4 __attribute__((ext_vector_type(4)));
typedef int          i32x4 __attribute__((ext_vector_type(4)));
typedef int          i32x8 __attribute__((ext_vector_type(8)));

#define B_ 8
#define T_ 2048
#define D_ 1024
#define H_ 64

union Frag { v16h v; v8h h[2]; };

__device__ __forceinline__ v8f wmma_f16(const Frag& a, const Frag& b, v8f c) {
  // D = A(16x32 f16) * B(32x16 f16) + C(16x16 f32)
  return __builtin_amdgcn_wmma_f32_16x16x32_f16(false, a.v, false, b.v,
                                                (short)0, c, false, false);
}

// xor-swizzle within 16-lane halves (matches C-layout row striping)
template<int XM>
__device__ __forceinline__ float swz_xor(float v) {
  int i = __builtin_bit_cast(int, v);
  i = __builtin_amdgcn_ds_swizzle(i, (XM << 10) | 0x1f);
  return __builtin_bit_cast(float, i);
}
__device__ __forceinline__ float redmax16(float v) {
  v = fmaxf(v, swz_xor<1>(v)); v = fmaxf(v, swz_xor<2>(v));
  v = fmaxf(v, swz_xor<4>(v)); v = fmaxf(v, swz_xor<8>(v));
  return v;
}
__device__ __forceinline__ float redsum16(float v) {
  v += swz_xor<1>(v); v += swz_xor<2>(v);
  v += swz_xor<4>(v); v += swz_xor<8>(v);
  return v;
}

// ---------------------------------------------------------------------------
// Tensor Data Mover: DMA a 2D f16 tile (tile1 rows x tile0 elems) from global
// into LDS, with per-row dword padding for bank-conflict-free reads.
// D# built per CDNA5 ISA 8.3/8.4 (group0: count/lds/global/type,
// group1: data_size, pad ctrl, dims, tile dims, dim0 stride).
// Toolchain builtin (clang-23): 6 args (g0 u32x4, g1 i32x8, g2 i32x4,
// g3 i32x4, extra i32x8, cpol i32).
// ---------------------------------------------------------------------------
__device__ __forceinline__ void tdm_load_tile_f16(
    const f16* lds_dst, const f16* gsrc,
    unsigned tdim0, unsigned tdim1,         // tensor extents (elements)
    unsigned tile0, unsigned tile1,         // tile extents (elements)
    unsigned stride0,                       // row stride in elements
    unsigned pad_int_code, unsigned pad_amt_code) {
#if __has_builtin(__builtin_amdgcn_tensor_load_to_lds)
  const unsigned long long ga = (unsigned long long)gsrc;
  const unsigned lds_off = (unsigned)(unsigned long long)lds_dst;  // low 32 = LDS byte offset
  u32x4 g0;
  g0[0] = 1u;                                   // count=1 (valid), user mode
  g0[1] = lds_off;                              // lds_addr
  g0[2] = (unsigned)ga;                         // global_addr[31:0]
  g0[3] = (unsigned)(ga >> 32) | (2u << 30);    // global_addr[56:32] | type=2
  i32x8 g1;
  g1[0] = (int)((1u << 16) |                    // data_size = 1 -> 2 bytes
                (1u << 20) |                    // pad_enable
                (pad_int_code << 22) |          // pad interval (2^(c+1) dwords)
                (pad_amt_code << 25));          // pad amount ((c+1) dwords)
  g1[1] = (int)(tdim0 << 16);                   // tensor_dim0[15:0]
  g1[2] = (int)((tdim0 >> 16) | (tdim1 << 16)); // tensor_dim0 hi | tensor_dim1 lo
  g1[3] = (int)((tdim1 >> 16) | (tile0 << 16)); // tensor_dim1 hi | tile_dim0
  g1[4] = (int)tile1;                           // tile_dim1 (tile_dim2=0)
  g1[5] = (int)stride0;                         // tensor_dim0_stride[31:0]
  g1[6] = 0;                                    // stride0 hi | stride1 lo
  g1[7] = 0;
  const i32x4 gz4 = {0, 0, 0, 0};
  const i32x8 gz8 = {0, 0, 0, 0, 0, 0, 0, 0};
  __builtin_amdgcn_tensor_load_to_lds(g0, g1, gz4, gz4, gz8, 0);
#else
  // Fallback: cooperative wave copy (wave32), row-major with padded stride.
  const int lane = (int)(threadIdx.x & 31);
  const unsigned row_halves = tile0 + 2 * (pad_amt_code + 1) * ((tile0 / 2) >> (pad_int_code + 1));
  for (unsigned r = 0; r < tile1; ++r)
    for (unsigned cx = (unsigned)lane * 8; cx < tile0; cx += 32 * 8)
      *(v8h*)((f16*)lds_dst + r * row_halves + cx) = *(const v8h*)(gsrc + r * stride0 + cx);
#endif
}

// ---------------------------------------------------------------------------
// Kernel 1: Q/K/V projection. Block = 256 thr (8 waves), 128 rows of x.
// ---------------------------------------------------------------------------
#define XSTR 40  // padded LDS row stride (halves) -> conflict-free ds_load_b128

__global__ __launch_bounds__(256)
void proj_qkv(const float* __restrict__ x,
              const float* __restrict__ Wq,
              const float* __restrict__ Wk,
              const float* __restrict__ Wv,
              f16* __restrict__ qg,   // [B*T][H]
              f16* __restrict__ kg,   // [B*T][H]
              f16* __restrict__ vtg)  // [B][H][T]
{
  __shared__ f16 xs[128 * XSTR];      // x tile  128 x 32 (f16)
  __shared__ f16 ws[3][64 * XSTR];    // W tiles transposed: [h][k] 64 x 32

  const int tid  = threadIdx.x;
  const int lane = tid & 31;
  const int wave = tid >> 5;
  const int m    = lane & 15;
  const int half = lane >> 4;
  const int rowbase = blockIdx.x * 128;

  const float* Wsrc[3] = {Wq, Wk, Wv};

  v8f acc[3][4];
#pragma unroll
  for (int w = 0; w < 3; ++w)
#pragma unroll
    for (int nt = 0; nt < 4; ++nt)
      acc[w][nt] = (v8f){0.f,0.f,0.f,0.f,0.f,0.f,0.f,0.f};

  for (int kt = 0; kt < D_; kt += 32) {
    // ---- stage x tile: 128x32 fp32 -> f16 LDS (+ prefetch next tile) ----
    {
      const int r0 = tid >> 3;            // 0..31
      const int c4 = (tid & 7) * 4;       // 0..28
#pragma unroll
      for (int p = 0; p < 4; ++p) {
        const int r = r0 + p * 32;
        const float* src = x + (size_t)(rowbase + r) * D_ + kt + c4;
        v4f xv = *(const v4f*)src;
        if (kt + 32 < D_) __builtin_prefetch(src + 32, 0, 0);  // global_prefetch_b8
        v4h hv; hv[0]=(f16)xv[0]; hv[1]=(f16)xv[1]; hv[2]=(f16)xv[2]; hv[3]=(f16)xv[3];
        *(v4h*)(&xs[r * XSTR + c4]) = hv;
      }
    }
    // ---- stage 3 weight tiles transposed: ws[w][h][k] ----
    {
      const int kr = tid >> 3;            // 0..31
      const int h0 = (tid & 7) * 8;       // 0..56
#pragma unroll
      for (int w = 0; w < 3; ++w) {
        const float* wp = Wsrc[w] + (size_t)(kt + kr) * H_ + h0;
        v4f a = *(const v4f*)(wp);
        v4f c = *(const v4f*)(wp + 4);
#pragma unroll
        for (int j = 0; j < 4; ++j) ws[w][(h0 + j) * XSTR + kr]     = (f16)a[j];
#pragma unroll
        for (int j = 0; j < 4; ++j) ws[w][(h0 + 4 + j) * XSTR + kr] = (f16)c[j];
      }
    }
    __syncthreads();

    // ---- A fragment (16 rows of this wave, K=32) ----
    Frag A;
    const f16* arow = &xs[(wave * 16 + m) * XSTR + half * 8];
    A.h[0] = *(const v8h*)(arow);        // K = half*8 + 0..7
    A.h[1] = *(const v8h*)(arow + 16);   // K = 16 + half*8 + 0..7

#pragma unroll
    for (int w = 0; w < 3; ++w)
#pragma unroll
      for (int nt = 0; nt < 4; ++nt) {
        Frag Bf;  // column n = nt*16+m, K = half*8 + 0..7, 16 + half*8 + 0..7
        const f16* brow = &ws[w][(nt * 16 + m) * XSTR + half * 8];
        Bf.h[0] = *(const v8h*)(brow);
        Bf.h[1] = *(const v8h*)(brow + 16);
        acc[w][nt] = wmma_f16(A, Bf, acc[w][nt]);
      }
    __syncthreads();
  }

  // ---- write back: Q,K row-major f16; V transposed [B][H][T] f16 ----
#pragma unroll
  for (int nt = 0; nt < 4; ++nt)
#pragma unroll
    for (int r = 0; r < 8; ++r) {
      const int row = rowbase + wave * 16 + r + 8 * half;
      const int h   = nt * 16 + m;
      qg[(size_t)row * H_ + h] = (f16)acc[0][nt][r];
      kg[(size_t)row * H_ + h] = (f16)acc[1][nt][r];
      const int bb = row >> 11;          // T_ = 2048
      const int tt = row & (T_ - 1);
      vtg[((size_t)bb * H_ + h) * T_ + tt] = (f16)acc[2][nt][r];
    }
}

// ---------------------------------------------------------------------------
// Kernel 2: flash attention. Block = 256 thr (8 waves) owns 128 consecutive
// queries; K/V chunks (32 keys) staged once per block by the TDM, double-
// buffered; each wave computes one 16-query tile.
// LDS row strides after TDM padding: K = 36 dwords (72 halves),
// V = 20 dwords (40 halves) -> conflict-free ds_load_b128 fragments.
// ---------------------------------------------------------------------------
#define KSTR 72
#define VSTR 40

__global__ __launch_bounds__(256)
void flash_attn(const f16* __restrict__ qg,
                const f16* __restrict__ kg,
                const f16* __restrict__ vtg,
                float* __restrict__ out)
{
  __shared__ f16 ksb[2][32 * KSTR];   // staged K chunk  [key][h]
  __shared__ f16 vsb[2][64 * VSTR];   // staged V^T chunk [h][key]
  __shared__ f16 pbuf[8][16 * 32];    // per-wave P staging (C->A layout)

  const int tid  = threadIdx.x;
  const int lane = tid & 31;
  const int wave = tid >> 5;
  const int m    = lane & 15;
  const int half = lane >> 4;
  const int b     = blockIdx.y;
  const int qbase = (blockIdx.x * 8 + wave) * 16;   // consecutive tiles per block

  // Q A-fragments for h=0..31 and h=32..63 (kept in registers)
  Frag Aq[2];
  {
    const f16* qrow = qg + ((size_t)b * T_ + qbase + m) * H_;
    Aq[0].h[0] = *(const v8h*)(qrow + half * 8);
    Aq[0].h[1] = *(const v8h*)(qrow + 16 + half * 8);
    Aq[1].h[0] = *(const v8h*)(qrow + 32 + half * 8);
    Aq[1].h[1] = *(const v8h*)(qrow + 48 + half * 8);
  }

  v8f O[4];
#pragma unroll
  for (int nt = 0; nt < 4; ++nt) O[nt] = (v8f){0.f,0.f,0.f,0.f,0.f,0.f,0.f,0.f};
  float Mrow[8], Lrow[8];
#pragma unroll
  for (int r = 0; r < 8; ++r) { Mrow[r] = -1e30f; Lrow[r] = 0.f; }

  const float SC2 = 0.125f * 1.44269504088896340736f;  // 1/sqrt(H) * log2(e)
  const int nch = (blockIdx.x * 128 + 112 + 47) >> 5;  // covers block's max query

  const f16* kcg = kg  + (size_t)b * T_ * H_;
  const f16* vcg = vtg + (size_t)b * H_ * T_;

  // prologue: TDM-stage chunk 0 into buffer 0 (one wave drives the DMA)
  if (wave == 0) {
    tdm_load_tile_f16(&ksb[0][0], kcg, 64, 32, 64, 32, H_, /*pad*/4, 3);
    tdm_load_tile_f16(&vsb[0][0], vcg, 32, 64, 32, 64, T_, /*pad*/3, 3);
    __builtin_amdgcn_s_wait_tensorcnt(0);
  }
  __syncthreads();

  for (int c = 0; c < nch; ++c) {
    const int kbase = c * 32;
    const int buf   = c & 1;

    // kick off next chunk's DMA while this one is consumed
    if (wave == 0 && c + 1 < nch) {
      const int nk = (c + 1) * 32;
      tdm_load_tile_f16(&ksb[buf ^ 1][0], kcg + (size_t)nk * H_, 64, 32, 64, 32, H_, 4, 3);
      tdm_load_tile_f16(&vsb[buf ^ 1][0], vcg + nk,              32, 64, 32, 64, T_, 3, 3);
    }

    if (kbase <= qbase + 15) {  // wave-uniform causal skip (EXEC stays full)
      // ---- S = Q K^T (two 16x16 tiles, contraction over H=64) ----
      v8f S[2];
#pragma unroll
      for (int s = 0; s < 2; ++s) {
        v8f ct = (v8f){0.f,0.f,0.f,0.f,0.f,0.f,0.f,0.f};
        const f16* krow = &ksb[buf][(s * 16 + m) * KSTR];
#pragma unroll
        for (int hs = 0; hs < 2; ++hs) {
          Frag Bk;  // column = key (n), contraction (h) at half*8 and 16 + half*8
          Bk.h[0] = *(const v8h*)(krow + hs * 32 + half * 8);
          Bk.h[1] = *(const v8h*)(krow + hs * 32 + 16 + half * 8);
          ct = wmma_f16(Aq[hs], Bk, ct);
        }
        S[s] = ct;
      }

      // ---- scale (log2 domain), causal mask, online softmax ----
      float tmax[8];
#pragma unroll
      for (int r = 0; r < 8; ++r) {
        const int qi = qbase + r + 8 * half;
        float s0 = (kbase + m      <= qi) ? S[0][r] * SC2 : -1e30f;
        float s1 = (kbase + 16 + m <= qi) ? S[1][r] * SC2 : -1e30f;
        S[0][r] = s0; S[1][r] = s1;
        tmax[r] = redmax16(fmaxf(s0, s1));
      }
      float corr[8], psum[8];
#pragma unroll
      for (int r = 0; r < 8; ++r) {
        const float mn = fmaxf(Mrow[r], tmax[r]);
        corr[r] = __builtin_amdgcn_exp2f(Mrow[r] - mn);
        Mrow[r] = mn;
        const float p0 = __builtin_amdgcn_exp2f(S[0][r] - mn);
        const float p1 = __builtin_amdgcn_exp2f(S[1][r] - mn);
        S[0][r] = p0; S[1][r] = p1;
        psum[r] = redsum16(p0 + p1);
      }
#pragma unroll
      for (int r = 0; r < 8; ++r) Lrow[r] = Lrow[r] * corr[r] + psum[r];
#pragma unroll
      for (int nt = 0; nt < 4; ++nt)
#pragma unroll
        for (int r = 0; r < 8; ++r) O[nt][r] *= corr[r];

      // ---- P: C-layout -> f16 A-layout via per-wave LDS round-trip ----
      f16* pb = pbuf[wave];
#pragma unroll
      for (int s = 0; s < 2; ++s)
#pragma unroll
        for (int r = 0; r < 8; ++r)
          pb[(r + 8 * half) * 32 + s * 16 + m] = (f16)S[s][r];
      asm volatile("s_wait_dscnt 0" ::: "memory");  // same-wave DS in-order

      Frag Ap;
      const f16* prow = pb + m * 32 + half * 8;
      Ap.h[0] = *(const v8h*)(prow);
      Ap.h[1] = *(const v8h*)(prow + 16);

      // ---- O += P V  (V^T staged: contiguous keys per h-row) ----
#pragma unroll
      for (int nt = 0; nt < 4; ++nt) {
        Frag Bv;
        const f16* vr = &vsb[buf][(nt * 16 + m) * VSTR + half * 8];
        Bv.h[0] = *(const v8h*)(vr);
        Bv.h[1] = *(const v8h*)(vr + 16);
        O[nt] = wmma_f16(Ap, Bv, O[nt]);
      }
    }

    if (wave == 0 && c + 1 < nch)
      __builtin_amdgcn_s_wait_tensorcnt(0);
    __syncthreads();
  }

  // ---- normalize and store fp32 output ----
  float inv[8];
#pragma unroll
  for (int r = 0; r < 8; ++r) inv[r] = __builtin_amdgcn_rcpf(Lrow[r]);
#pragma unroll
  for (int nt = 0; nt < 4; ++nt)
#pragma unroll
    for (int r = 0; r < 8; ++r) {
      const int row = qbase + r + 8 * half;
      out[((size_t)b * T_ + row) * H_ + nt * 16 + m] = O[nt][r] * inv[r];
    }
}

// ---------------------------------------------------------------------------
extern "C" void kernel_launch(void* const* d_in, const int* in_sizes, int n_in,
                              void* d_out, int out_size, void* d_ws, size_t ws_size,
                              hipStream_t stream) {
  const float* x  = (const float*)d_in[0];
  const float* Wq = (const float*)d_in[1];
  const float* Wk = (const float*)d_in[2];
  const float* Wv = (const float*)d_in[3];
  float* out = (float*)d_out;

  // workspace: Q, K, V^T as f16 (2 MB each)
  f16* qg  = (f16*)d_ws;
  f16* kg  = qg + (size_t)B_ * T_ * H_;
  f16* vtg = kg + (size_t)B_ * T_ * H_;

  proj_qkv<<<dim3((B_ * T_) / 128), 256, 0, stream>>>(x, Wq, Wk, Wv, qg, kg, vtg);
  flash_attn<<<dim3(T_ / 16 / 8, B_), 256, 0, stream>>>(qg, kg, vtg, out);
}
